// ANI2xCore_21955872817671
// MI455X (gfx1250) — compile-verified, matrix operand order checked by reading, not run on hardware
//
#include <hip/hip_runtime.h>

// ---------------- types ----------------
typedef _Float16 v16h __attribute__((ext_vector_type(16)));
typedef _Float16 v8h  __attribute__((ext_vector_type(8)));
typedef _Float16 v4h  __attribute__((ext_vector_type(4)));
typedef float    v8f  __attribute__((ext_vector_type(8)));
typedef float    v4f  __attribute__((ext_vector_type(4)));

#define NATOMS 65536
#define AEVD   1008
#define NSPEC  7
#define MTILE  32

// layer widths per expert: [in, l1, l2, l3, 1]
__constant__ int LAYD[7][5] = {
    {1008, 256, 192, 160, 1},  // H
    {1008, 224, 192, 160, 1},  // C
    {1008, 192, 160, 128, 1},  // O
    {1008, 192, 160, 128, 1},  // N
    {1008, 160, 128,  96, 1},  // S
    {1008, 160, 128,  96, 1},  // F
    {1008, 160, 128,  96, 1},  // Cl
};

struct WPtrs { const float* W[28]; };
struct BPtrs { const float* B[28]; };

// ws layout (bytes):
//   [0,1024)          : int meta: [0..6]=counts [8..15]=offsets [16..22]=cursor
//   [1024, 263168)    : int idx[65536]  (atoms bucketed by species)
//   [263168, ...)     : _Float16 packed weights (~3.5 MB), B-fragment layout
#define OFF_IDX 1024
#define OFF_W   263168

__device__ __forceinline__ int pad16(int x) { return (x + 15) & ~15; }
__device__ __forceinline__ int layKd(int e, int l) { return (l == 0) ? 1024 : LAYD[e][l]; }
__device__ __forceinline__ int layNd(int e, int l) { return pad16(LAYD[e][l + 1]); }
__device__ int layOff(int li) {            // f16 elements before layer li (li = e*4+l)
    int off = 0;
    for (int i = 0; i < li; ++i) off += layKd(i >> 2, i & 3) * layNd(i >> 2, i & 3);
    return off;
}
// K position inside a 32-deep fragment for v16h element `e`, lane-half `g`
// (per CDNA5 ISA 7.12.2 16-bit operand layout)
__device__ __forceinline__ int kmap(int e, int g) {
    int v = e >> 1, h = e & 1;
    return (v < 4) ? (2 * v + h + 8 * g) : (2 * (v - 4) + h + 16 + 8 * g);
}
__device__ __forceinline__ float celu01(float x) {
    return (x > 0.f) ? x : 0.1f * (__expf(x * 10.f) - 1.f);
}

// ---------------- routing kernels ----------------
__global__ void k_init(int* meta) {
    int t = threadIdx.x;
    if (t < 24) meta[t] = 0;
}

__global__ void k_hist(const int* __restrict__ sp, int* __restrict__ meta) {
    __shared__ int lc[NSPEC];
    if (threadIdx.x < NSPEC) lc[threadIdx.x] = 0;
    __syncthreads();
    int i = blockIdx.x * blockDim.x + threadIdx.x;
    if (i < NATOMS) atomicAdd(&lc[sp[i]], 1);
    __syncthreads();
    if (threadIdx.x < NSPEC) atomicAdd(&meta[threadIdx.x], lc[threadIdx.x]);
}

__global__ void k_offsets(int* meta) {
    if (threadIdx.x == 0 && blockIdx.x == 0) {
        int acc = 0;
        for (int s = 0; s < NSPEC; ++s) {
            meta[8 + s]  = acc;   // offsets
            meta[16 + s] = acc;   // cursor
            acc += meta[s];
        }
        meta[8 + NSPEC] = acc;    // == NATOMS
    }
}

__global__ void k_scatter(const int* __restrict__ sp, int* __restrict__ meta,
                          int* __restrict__ idx) {
    int i = blockIdx.x * blockDim.x + threadIdx.x;
    if (i < NATOMS) {
        int s = sp[i];
        int p = atomicAdd(&meta[16 + s], 1);
        idx[p] = i;
    }
}

// ---------------- weight pack: f32 (in,out) -> f16 WMMA-B fragments ----------------
__global__ void k_pack(WPtrs P, _Float16* __restrict__ wbuf) {
    int li = blockIdx.y;            // expert*4 + layer
    int e = li >> 2, l = li & 3;
    int Kd = layKd(e, l), Nd = layNd(e, l);
    int sz = Kd * Nd;
    int i = blockIdx.x * blockDim.x + threadIdx.x;
    if (i >= sz) return;
    int f    = i & 511;             // element within one 32x16 fragment
    int blk  = i >> 9;              // fragment index (kt * NT + nt)
    int NT   = Nd >> 4;
    int kt   = blk / NT, nt = blk - kt * NT;
    int lane = f >> 4, el = f & 15;
    int g    = lane >> 4;
    int K    = kt * 32 + kmap(el, g);
    int N    = nt * 16 + (lane & 15);
    int in_d = LAYD[e][l], out_d = LAYD[e][l + 1];
    float w = (K < in_d && N < out_d) ? P.W[li][(size_t)K * out_d + N] : 0.0f;
    wbuf[(size_t)layOff(li) + i] = (_Float16)w;
}

// ---------------- fused expert MLP (32-atom tile, 8 waves, WMMA) ----------------
__global__ __launch_bounds__(256)
void k_mlp(const float* __restrict__ aev, const int* __restrict__ meta,
           const int* __restrict__ idx, const _Float16* __restrict__ wbuf,
           BPtrs BP, float* __restrict__ out) {
    const int s     = blockIdx.y;
    const int begin = meta[8 + s];
    const int endv  = meta[8 + s + 1];
    const int start = begin + (int)blockIdx.x * MTILE;
    if (start >= endv) return;

    // K=1024 padded AEV tile, pitch 1032 (+16B pad -> conflict-free ds_load_b128)
    __shared__ __attribute__((aligned(16))) _Float16 bufA[MTILE * 1032];  // 66.0 KB
    __shared__ __attribute__((aligned(16))) _Float16 buf1[MTILE * 264];   // 16.9 KB
    __shared__ __attribute__((aligned(16))) _Float16 buf2[MTILE * 200];   // 12.8 KB
    __shared__ int aidx[MTILE];

    const int tid  = threadIdx.x;
    const int lane = tid & 31, wv = tid >> 5;
    const int g    = lane >> 4, nl = lane & 15;

    if (tid < MTILE) aidx[tid] = (start + tid < endv) ? idx[start + tid] : -1;
    __syncthreads();

    // Stage AEV tile: non-temporal f32 stream -> f16 LDS (keep weights hot in L2)
    for (int t = tid; t < MTILE * 252; t += 256) {
        int m = t / 252, c = t - m * 252;
        v4f val = {0.f, 0.f, 0.f, 0.f};
        int a = aidx[m];
        if (a >= 0)
            val = __builtin_nontemporal_load((const v4f*)(aev + (size_t)a * AEVD) + c);
        v4h hv = {(_Float16)val[0], (_Float16)val[1], (_Float16)val[2], (_Float16)val[3]};
        *(v4h*)&bufA[m * 1032 + c * 4] = hv;
    }
    for (int t = tid; t < MTILE * 24; t += 256) {    // zero K padding 1008..1031
        int m = t / 24, k2 = AEVD + (t - m * 24);
        bufA[m * 1032 + k2] = (_Float16)0.f;
    }
    __syncthreads();

    // per-layer packed weight offsets for this expert
    int wo[4];
    {
        int off = layOff(s * 4);
        for (int l = 0; l < 4; ++l) { wo[l] = off; off += layKd(s, l) * layNd(s, l); }
    }

    _Float16* src = bufA;
    int PS = 1032;
    for (int l = 0; l < 4; ++l) {
        const int outd = LAYD[s][l + 1];
        const int Kd   = (l == 0) ? 1024 : LAYD[s][l];
        const int Nd   = pad16(outd);
        const int PD   = Nd + 8;
        _Float16* dst  = (l == 0) ? buf1 : (l == 1) ? buf2 : buf1;
        const _Float16* W = wbuf + wo[l];
        const float* bias = BP.B[s * 4 + l];
        const int NT = Nd >> 4, KT = Kd >> 5;

        // wave wv owns column tiles nt0 = wv and nt1 = wv + 8 (NT <= 16 always);
        // both 16-row A tiles (mt = 0,1) share every B fragment -> up to 4 WMMA/K-step
        const int  nt0  = wv, nt1 = wv + 8;
        const bool has1 = (nt1 < NT);
        if (nt0 < NT) {
            const int n00 = nt0 << 4, n01 = nt1 << 4;
            float bv0 = (n00 + nl < outd) ? bias[n00 + nl] : 0.0f;
            float bv1 = has1 ? ((n01 + nl < outd) ? bias[n01 + nl] : 0.0f) : 0.0f;
            v8f a00, a10, a01, a11;
            for (int r = 0; r < 8; ++r) {            // fold bias into C
                a00[r] = bv0; a10[r] = bv0; a01[r] = bv1; a11[r] = bv1;
            }

            const _Float16* ap0 = &src[nl * PS + 8 * g];          // rows 0..15
            const _Float16* ap1 = &src[(nl + 16) * PS + 8 * g];   // rows 16..31
            const _Float16* wp0 = W + ((size_t)nt0 << 9) + lane * 16;
            const _Float16* wp1 = W + ((size_t)nt1 << 9) + lane * 16;
            const size_t wstep  = (size_t)NT << 9;

            if (has1) {
                for (int kt = 0; kt < KT; ++kt) {
                    const int k0 = kt << 5;
                    v8h lo0 = *(const v8h*)(ap0 + k0);
                    v8h hi0 = *(const v8h*)(ap0 + k0 + 16);
                    v8h lo1 = *(const v8h*)(ap1 + k0);
                    v8h hi1 = *(const v8h*)(ap1 + k0 + 16);
                    v16h A0 = __builtin_shufflevector(lo0, hi0, 0, 1, 2, 3, 4, 5, 6, 7,
                                                      8, 9, 10, 11, 12, 13, 14, 15);
                    v16h A1 = __builtin_shufflevector(lo1, hi1, 0, 1, 2, 3, 4, 5, 6, 7,
                                                      8, 9, 10, 11, 12, 13, 14, 15);
                    v16h B0 = *(const v16h*)wp0;  wp0 += wstep;
                    v16h B1 = *(const v16h*)wp1;  wp1 += wstep;
                    a00 = __builtin_amdgcn_wmma_f32_16x16x32_f16(false, A0, false, B0,
                                                                 (short)0, a00, false, false);
                    a10 = __builtin_amdgcn_wmma_f32_16x16x32_f16(false, A1, false, B0,
                                                                 (short)0, a10, false, false);
                    a01 = __builtin_amdgcn_wmma_f32_16x16x32_f16(false, A0, false, B1,
                                                                 (short)0, a01, false, false);
                    a11 = __builtin_amdgcn_wmma_f32_16x16x32_f16(false, A1, false, B1,
                                                                 (short)0, a11, false, false);
                }
            } else {
                for (int kt = 0; kt < KT; ++kt) {
                    const int k0 = kt << 5;
                    v8h lo0 = *(const v8h*)(ap0 + k0);
                    v8h hi0 = *(const v8h*)(ap0 + k0 + 16);
                    v8h lo1 = *(const v8h*)(ap1 + k0);
                    v8h hi1 = *(const v8h*)(ap1 + k0 + 16);
                    v16h A0 = __builtin_shufflevector(lo0, hi0, 0, 1, 2, 3, 4, 5, 6, 7,
                                                      8, 9, 10, 11, 12, 13, 14, 15);
                    v16h A1 = __builtin_shufflevector(lo1, hi1, 0, 1, 2, 3, 4, 5, 6, 7,
                                                      8, 9, 10, 11, 12, 13, 14, 15);
                    v16h B0 = *(const v16h*)wp0;  wp0 += wstep;
                    a00 = __builtin_amdgcn_wmma_f32_16x16x32_f16(false, A0, false, B0,
                                                                 (short)0, a00, false, false);
                    a10 = __builtin_amdgcn_wmma_f32_16x16x32_f16(false, A1, false, B0,
                                                                 (short)0, a10, false, false);
                }
            }

            if (l < 3) {
                // CELU(0.1) + f16 back to LDS (D layout: lane=N, vgpr r -> M=r+8g)
                for (int r = 0; r < 8; ++r) {
                    int row = r + 8 * g;
                    dst[row * PD + n00 + nl]        = (_Float16)celu01(a00[r]);
                    dst[(row + 16) * PD + n00 + nl] = (_Float16)celu01(a10[r]);
                    if (has1) {
                        dst[row * PD + n01 + nl]        = (_Float16)celu01(a01[r]);
                        dst[(row + 16) * PD + n01 + nl] = (_Float16)celu01(a11[r]);
                    }
                }
            } else if (nl == 0) {
                // final Linear(->1): scatter f32 energies for all 32 atoms
                for (int r = 0; r < 8; ++r) {
                    int row = r + 8 * g;
                    if (aidx[row] >= 0)      out[aidx[row]]      = a00[r];
                    if (aidx[row + 16] >= 0) out[aidx[row + 16]] = a10[r];
                }
            }
        }
        __syncthreads();
        src = dst;
        PS  = PD;
    }
}

// ---------------- launch ----------------
extern "C" void kernel_launch(void* const* d_in, const int* in_sizes, int n_in,
                              void* d_out, int out_size, void* d_ws, size_t ws_size,
                              hipStream_t stream) {
    const int*   species = (const int*)d_in[0];
    const float* aev     = (const float*)d_in[1];

    WPtrs wp; BPtrs bp;
    for (int t = 0; t < 28; ++t) {          // params flatten: (W,b) per layer per expert
        wp.W[t] = (const float*)d_in[2 + 2 * t];
        bp.B[t] = (const float*)d_in[3 + 2 * t];
    }

    char*     ws   = (char*)d_ws;
    int*      meta = (int*)ws;
    int*      idx  = (int*)(ws + OFF_IDX);
    _Float16* wbuf = (_Float16*)(ws + OFF_W);

    k_init   <<<1, 32, 0, stream>>>(meta);
    k_hist   <<<NATOMS / 256, 256, 0, stream>>>(species, meta);
    k_offsets<<<1, 1, 0, stream>>>(meta);
    k_scatter<<<NATOMS / 256, 256, 0, stream>>>(species, meta, idx);
    k_pack   <<<dim3(1024, 28), 256, 0, stream>>>(wp, wbuf);
    k_mlp    <<<dim3(NATOMS / MTILE, NSPEC), 256, 0, stream>>>(aev, meta, idx, wbuf, bp,
                                                               (float*)d_out);
}
